// SingleStreamBlock_33105607917601
// MI455X (gfx1250) — compile-verified, matrix operand order checked by reading, not run on hardware
//
#include <hip/hip_runtime.h>
#include <hip/hip_bf16.h>
#include <math.h>

// ---------------- constants ----------------
#define L_SEQ 2048
#define NHEAD 24
#define DHEAD 128
#define HID   3072
#define MLPD  12288
#define N1    21504     // 3*HID + MLP
#define K2    15360     // HID + MLP
#define EPSF  1e-6f

typedef __bf16 bf16_t;
typedef bf16_t bf16x16 __attribute__((ext_vector_type(16)));
typedef float  f32x8  __attribute__((ext_vector_type(8)));
typedef unsigned int u32x4 __attribute__((ext_vector_type(4)));
typedef int          i32x8 __attribute__((ext_vector_type(8)));
typedef int          i32x4 __attribute__((ext_vector_type(4)));

union Frag {
    bf16x16        bf;
    unsigned int   u32[8];
    unsigned short u16[16];
};

__device__ __forceinline__ unsigned short f32_to_bf16_bits(float f) {
    union { float f; unsigned int u; } v; v.f = f;
    unsigned int u = v.u;
    u += 0x7FFFu + ((u >> 16) & 1u);   // round-to-nearest-even
    return (unsigned short)(u >> 16);
}

__device__ __forceinline__ float gelu_tanh(float v) {
    return 0.5f * v * (1.f + tanhf(0.7978845608028654f * (v + 0.044715f * v * v * v)));
}

// ------------------------------------------------------------
// TDM: async 2D tile load (bf16 elements) global -> LDS.
// Descriptor per CDNA5 ISA.
// dim0   = tensor length along contiguous axis (elements)
// dim1   = tensor length along row axis
// stride0= elements between consecutive rows
// tile0  = tile width (contiguous elements), tile1 = tile rows
// Uses the 6-arg (clang-23 / therock-10.0) builtin form.
// ------------------------------------------------------------
__device__ __forceinline__ void tdm_load_2d_bf16(const void* gsrc, unsigned lds_off,
                                                 unsigned dim0, unsigned dim1,
                                                 unsigned stride0,
                                                 unsigned tile0, unsigned tile1) {
    unsigned long long ga = (unsigned long long)(uintptr_t)gsrc;
    u32x4 g0;
    g0[0] = 1u;                                   // count=1 (valid user descriptor)
    g0[1] = lds_off;                              // lds_addr[63:32]
    g0[2] = (unsigned)ga;                         // global_addr[95:64]
    g0[3] = (unsigned)(ga >> 32) | (2u << 30);    // global_addr hi + type=2 (image)
    i32x8 g1;
    g1[0] = 1 << 16;                              // workgroup_mask=0, data_size=1 (2B)
    g1[1] = (int)((dim0 & 0xFFFFu) << 16);        // tensor_dim0[15:0] @ [63:48]
    g1[2] = (int)((dim0 >> 16) | ((dim1 & 0xFFFFu) << 16)); // dim0 hi | dim1 lo
    g1[3] = (int)((dim1 >> 16) | (tile0 << 16));  // dim1 hi | tile_dim0
    g1[4] = (int)tile1;                           // tile_dim1 (tile_dim2=0)
    g1[5] = (int)stride0;                         // tensor_dim0_stride lo32
    g1[6] = 0;
    g1[7] = 0;
    i32x4 z4 = {0, 0, 0, 0};                      // groups 2/3 unused (2D tensor)
    i32x8 z8 = {0, 0, 0, 0, 0, 0, 0, 0};
    __builtin_amdgcn_tensor_load_to_lds(g0, g1, z4, z4, z8, 0);
}

// ============================================================
// Kernel 1: mod = silu(vec) @ mod_w + mod_b      (1x3072 @ 3072x9216)
// ============================================================
__global__ void mod_gemv_kernel(const float* __restrict__ vec,
                                const float* __restrict__ mod_w,
                                const float* __restrict__ mod_b,
                                float* __restrict__ mod) {
    __shared__ float sv[HID];
    for (int i = threadIdx.x; i < HID; i += blockDim.x) {
        float x = vec[i];
        sv[i] = x / (1.f + __expf(-x));
    }
    __syncthreads();
    int j = blockIdx.x * blockDim.x + threadIdx.x;
    if (j >= 3 * HID) return;
    float acc = mod_b[j];
    for (int i = 0; i < HID; ++i)
        acc += sv[i] * mod_w[i * (3 * HID) + j];
    mod[j] = acc;
}

// ============================================================
// Kernel 2: LayerNorm + modulate -> x_mod (bf16 bits)
// ============================================================
__global__ void ln_mod_kernel(const float* __restrict__ x,
                              const float* __restrict__ mod,   // [shift|scale|gate]
                              const float* __restrict__ gamma,
                              const float* __restrict__ beta,
                              unsigned short* __restrict__ xmod) {
    const int row = blockIdx.x;
    const float* xr = x + (size_t)row * HID;
    const int lane = threadIdx.x & 31, w = threadIdx.x >> 5;
    float s = 0.f, s2 = 0.f;
    for (int i = threadIdx.x; i < HID; i += 256) {
        float v = xr[i]; s += v; s2 += v * v;
    }
    #pragma unroll
    for (int off = 16; off; off >>= 1) { s += __shfl_xor(s, off); s2 += __shfl_xor(s2, off); }
    __shared__ float rs[8], rs2[8];
    if (lane == 0) { rs[w] = s; rs2[w] = s2; }
    __syncthreads();
    float S = 0.f, S2 = 0.f;
    #pragma unroll
    for (int i = 0; i < 8; ++i) { S += rs[i]; S2 += rs2[i]; }
    const float mu  = S * (1.f / HID);
    const float var = S2 * (1.f / HID) - mu * mu;
    const float inv = rsqrtf(var + EPSF);
    for (int i = threadIdx.x; i < HID; i += 256) {
        float ln = (xr[i] - mu) * inv * gamma[i] + beta[i];
        float v  = (1.f + mod[HID + i]) * ln + mod[i];
        xmod[(size_t)row * HID + i] = f32_to_bf16_bits(v);
    }
}

// ============================================================
// Shared GEMM core: C(128x128) = A_bf16(MxK) * B_f32->bf16(KxN)
// A tile staged by TDM (tensor_load_to_lds); B tile converted+transposed
// by VALU. 8 waves; wave (wm,wn) owns 32x64 (2x4 WMMA tiles).
// ============================================================
__device__ __forceinline__ void gemm_core(const unsigned short* __restrict__ A, int lda,
                                          const float* __restrict__ B, int ldb,
                                          int K, int rowsA, int m0, int n0,
                                          unsigned short* lA, unsigned short* lB,
                                          f32x8 acc[2][4]) {
    const int tid = threadIdx.x;
    const int lane = tid & 31, w = tid >> 5;
    const int wm = w & 3, wn = w >> 2;
    const int h2 = lane >> 4, ln = lane & 15;
    const unsigned lA_off = (unsigned)(uintptr_t)lA;   // low 32 bits = LDS byte offset

    #pragma unroll
    for (int t = 0; t < 2; ++t)
        #pragma unroll
        for (int u = 0; u < 4; ++u)
            #pragma unroll
            for (int e = 0; e < 8; ++e) acc[t][u][e] = 0.f;

    for (int k0 = 0; k0 < K; k0 += 32) {
        __syncthreads();
        if (tid < 32) {    // wave 0 issues the A-tile DMA: 128 rows x 32 bf16
            tdm_load_2d_bf16(A + (size_t)m0 * lda + k0, lA_off,
                             (unsigned)K, (unsigned)rowsA, (unsigned)lda,
                             32u, 128u);
            __builtin_amdgcn_s_wait_tensorcnt(0);
        }
        {   // stage B tile transposed to [n][k], converting f32 -> bf16
            int kr = tid >> 3, ns = (tid & 7) * 16;
            const float* src = B + (size_t)(k0 + kr) * ldb + n0 + ns;
            #pragma unroll
            for (int e = 0; e < 16; ++e)
                lB[(ns + e) * 32 + kr] = f32_to_bf16_bits(src[e]);
        }
        __syncthreads();
        if (k0 + 32 < K)   // pull next weight rows toward L2 (global_prefetch_b8)
            __builtin_prefetch(B + (size_t)(k0 + 32 + (tid >> 3)) * ldb + n0, 0, 0);

        // preload all fragments, then burst 8 back-to-back WMMAs
        Frag a[2], b[4];
        #pragma unroll
        for (int t = 0; t < 2; ++t) {
            int m = wm * 32 + t * 16 + ln;
            #pragma unroll
            for (int j = 0; j < 8; ++j) {
                int kk = (j >> 2) * 16 + h2 * 8 + (j & 3) * 2;
                a[t].u32[j] = *(const unsigned int*)(lA + m * 32 + kk);
            }
        }
        #pragma unroll
        for (int u = 0; u < 4; ++u) {
            int n = wn * 64 + u * 16 + ln;
            #pragma unroll
            for (int j = 0; j < 8; ++j) {
                int kk = (j >> 2) * 16 + h2 * 8 + (j & 3) * 2;
                b[u].u32[j] = *(const unsigned int*)(lB + n * 32 + kk);
            }
        }
        #pragma unroll
        for (int u = 0; u < 4; ++u)
            #pragma unroll
            for (int t = 0; t < 2; ++t)
                acc[t][u] = __builtin_amdgcn_wmma_f32_16x16x32_bf16(
                    false, a[t].bf, false, b[u].bf, (short)0, acc[t][u], false, false);
    }
}

// ============================================================
// Kernel 3: h = x_mod @ w1 + b1 ; split qkv(f32) / gelu(mlp)->A2(bf16)
// ============================================================
__global__ void gemm1_kernel(const unsigned short* __restrict__ xmod,
                             const float* __restrict__ w1,
                             const float* __restrict__ b1,
                             float* __restrict__ qkv,
                             unsigned short* __restrict__ a2) {
    __shared__ unsigned short lA[128 * 32];
    __shared__ unsigned short lB[128 * 32];
    const int m0 = blockIdx.y * 128, n0 = blockIdx.x * 128;
    f32x8 acc[2][4];
    gemm_core(xmod, HID, w1, N1, HID, L_SEQ, m0, n0, lA, lB, acc);

    const int lane = threadIdx.x & 31, w = threadIdx.x >> 5;
    const int wm = w & 3, wn = w >> 2, h2 = lane >> 4, ln = lane & 15;
    #pragma unroll
    for (int t = 0; t < 2; ++t)
        #pragma unroll
        for (int u = 0; u < 4; ++u)
            #pragma unroll
            for (int r = 0; r < 8; ++r) {
                int m = m0 + wm * 32 + t * 16 + r + 8 * h2;
                int n = n0 + wn * 64 + u * 16 + ln;
                float v = acc[t][u][r] + b1[n];
                if (n < 3 * HID) {
                    qkv[(size_t)m * (3 * HID) + n] = v;
                } else {
                    a2[(size_t)m * K2 + HID + (n - 3 * HID)] = f32_to_bf16_bits(gelu_tanh(v));
                }
            }
}

// ============================================================
// Kernel 4: rmsnorm(q,k)+rope, v passthrough -> bf16 [H][L][D]
// One wave per (head, token) row.
// ============================================================
__global__ void qkv_prep_kernel(const float* __restrict__ qkv,
                                const float* __restrict__ pe,
                                const float* __restrict__ q_scale,
                                const float* __restrict__ k_scale,
                                unsigned short* __restrict__ qb,
                                unsigned short* __restrict__ kb,
                                unsigned short* __restrict__ vb) {
    const int w = threadIdx.x >> 5, lane = threadIdx.x & 31;
    const int gw = blockIdx.x * 8 + w;
    const int head = gw >> 11;          // / 2048
    const int l = gw & 2047;
    const float* row = qkv + (size_t)l * (3 * HID) + head * DHEAD;
    const int d0 = lane * 4;

    float q4[4], k4[4], v4[4];
    float sq = 0.f, sk = 0.f;
    #pragma unroll
    for (int e = 0; e < 4; ++e) {
        q4[e] = row[d0 + e];
        k4[e] = row[HID + d0 + e];
        v4[e] = row[2 * HID + d0 + e];
        sq += q4[e] * q4[e];
        sk += k4[e] * k4[e];
    }
    #pragma unroll
    for (int off = 16; off; off >>= 1) { sq += __shfl_xor(sq, off); sk += __shfl_xor(sk, off); }
    const float rq = rsqrtf(sq * (1.f / DHEAD) + EPSF);
    const float rk = rsqrtf(sk * (1.f / DHEAD) + EPSF);

    const size_t obase = ((size_t)head * L_SEQ + l) * DHEAD + d0;
    const float* per = pe + (size_t)l * 256;    // [64][2][2] per token
    #pragma unroll
    for (int p = 0; p < 2; ++p) {
        int e = p * 2;
        int i = (d0 + e) >> 1;
        float a0 = q4[e]     * rq * q_scale[d0 + e];
        float a1 = q4[e + 1] * rq * q_scale[d0 + e + 1];
        float b0 = k4[e]     * rk * k_scale[d0 + e];
        float b1v = k4[e + 1] * rk * k_scale[d0 + e + 1];
        float p00 = per[i * 4 + 0], p01 = per[i * 4 + 1];
        float p10 = per[i * 4 + 2], p11 = per[i * 4 + 3];
        qb[obase + e]     = f32_to_bf16_bits(p00 * a0 + p01 * a1);
        qb[obase + e + 1] = f32_to_bf16_bits(p10 * a0 + p11 * a1);
        kb[obase + e]     = f32_to_bf16_bits(p00 * b0 + p01 * b1v);
        kb[obase + e + 1] = f32_to_bf16_bits(p10 * b0 + p11 * b1v);
    }
    #pragma unroll
    for (int e = 0; e < 4; ++e) vb[obase + e] = f32_to_bf16_bits(v4[e]);
}

// ============================================================
// Kernel 5: flash attention, bf16 WMMA. Block = (head, 128 q rows),
// 8 waves x 16 q-rows each; key tiles of 32 staged in LDS
// (K tile via TDM, V tile transposed by VALU).
// ============================================================
__global__ void attn_kernel(const unsigned short* __restrict__ qb,
                            const unsigned short* __restrict__ kb,
                            const unsigned short* __restrict__ vb,
                            unsigned short* __restrict__ a2) {
    __shared__ unsigned short lK[32 * 128];    // [key][d]
    __shared__ unsigned short lVt[128 * 32];   // [d][key]
    __shared__ unsigned short lP[8][16 * 32];  // per-wave P scratch
    const int tid = threadIdx.x, lane = tid & 31, w = tid >> 5;
    const int h2 = lane >> 4, ln = lane & 15;
    const int head = blockIdx.x >> 4;
    const int qrow0 = (blockIdx.x & 15) * 128 + w * 16;
    const float sm = 0.08838834764831845f;     // 1/sqrt(128)
    const unsigned lK_off = (unsigned)(uintptr_t)lK;

    // preload Q fragments (16 rows x 128 d = 4 A-frags)
    Frag qf[4];
    #pragma unroll
    for (int c = 0; c < 4; ++c)
        #pragma unroll
        for (int j = 0; j < 8; ++j) {
            int d = c * 32 + (j >> 2) * 16 + h2 * 8 + (j & 3) * 2;
            qf[c].u32[j] = *(const unsigned int*)(qb +
                ((size_t)(head * L_SEQ + qrow0 + ln)) * DHEAD + d);
        }

    f32x8 o[8];
    #pragma unroll
    for (int u = 0; u < 8; ++u)
        #pragma unroll
        for (int e = 0; e < 8; ++e) o[u][e] = 0.f;
    float rmax[8], rsum[8];
    #pragma unroll
    for (int r = 0; r < 8; ++r) { rmax[r] = -3.0e38f; rsum[r] = 0.f; }

    for (int k0 = 0; k0 < L_SEQ; k0 += 32) {
        __syncthreads();
        if (tid < 32) {   // wave 0: DMA K tile (32 keys x 128 d, contiguous rows)
            tdm_load_2d_bf16(kb + ((size_t)(head * L_SEQ + k0)) * DHEAD, lK_off,
                             (unsigned)DHEAD, (unsigned)(NHEAD * L_SEQ), (unsigned)DHEAD,
                             128u, 32u);
            __builtin_amdgcn_s_wait_tensorcnt(0);
        }
        {   // stage V transposed [d][key]
            int key = tid >> 3, ds = (tid & 7) * 16;
            const unsigned short* vs = vb + ((size_t)(head * L_SEQ + k0 + key)) * DHEAD + ds;
            #pragma unroll
            for (int e = 0; e < 16; ++e) lVt[(ds + e) * 32 + key] = vs[e];
        }
        __syncthreads();

        // scores S = Q K^T  (16 x 32 keys), 2 N-tiles x 4 d-chunks
        f32x8 s[2];
        #pragma unroll
        for (int u = 0; u < 2; ++u)
            #pragma unroll
            for (int e = 0; e < 8; ++e) s[u][e] = 0.f;
        #pragma unroll
        for (int u = 0; u < 2; ++u)
            #pragma unroll
            for (int c = 0; c < 4; ++c) {
                Frag b;
                int key = u * 16 + ln;
                #pragma unroll
                for (int j = 0; j < 8; ++j) {
                    int dd = c * 32 + (j >> 2) * 16 + h2 * 8 + (j & 3) * 2;
                    b.u32[j] = *(const unsigned int*)(lK + key * 128 + dd);
                }
                s[u] = __builtin_amdgcn_wmma_f32_16x16x32_bf16(
                    false, qf[c].bf, false, b.bf, (short)0, s[u], false, false);
            }

        // online softmax (row = r + 8*h2; 16 cols of a tile live in the 16-lane half)
        #pragma unroll
        for (int r = 0; r < 8; ++r) {
            float mx = fmaxf(s[0][r], s[1][r]) * sm;
            #pragma unroll
            for (int off = 1; off < 16; off <<= 1) mx = fmaxf(mx, __shfl_xor(mx, off));
            float mnew = fmaxf(rmax[r], mx);
            float alpha = __expf(rmax[r] - mnew);
            rmax[r] = mnew;
            float p0 = __expf(s[0][r] * sm - mnew);
            float p1 = __expf(s[1][r] * sm - mnew);
            s[0][r] = p0; s[1][r] = p1;
            float ps = p0 + p1;
            #pragma unroll
            for (int off = 1; off < 16; off <<= 1) ps += __shfl_xor(ps, off);
            rsum[r] = rsum[r] * alpha + ps;
            #pragma unroll
            for (int u = 0; u < 8; ++u) o[u][r] *= alpha;
        }

        // P (C-layout) -> per-wave LDS -> A-fragment layout
        unsigned short* lp = lP[w];
        #pragma unroll
        for (int u = 0; u < 2; ++u)
            #pragma unroll
            for (int r = 0; r < 8; ++r)
                lp[(r + 8 * h2) * 32 + u * 16 + ln] = f32_to_bf16_bits(s[u][r]);
        Frag pa;
        #pragma unroll
        for (int j = 0; j < 8; ++j) {
            int kk = (j >> 2) * 16 + h2 * 8 + (j & 3) * 2;
            pa.u32[j] = *(const unsigned int*)(lp + ln * 32 + kk);
        }

        // O += P V  (K-dim = 32 keys, 8 d-tiles)
        #pragma unroll
        for (int u = 0; u < 8; ++u) {
            Frag b;
            int dd = u * 16 + ln;
            #pragma unroll
            for (int j = 0; j < 8; ++j) {
                int key = (j >> 2) * 16 + h2 * 8 + (j & 3) * 2;
                b.u32[j] = *(const unsigned int*)(lVt + dd * 32 + key);
            }
            o[u] = __builtin_amdgcn_wmma_f32_16x16x32_bf16(
                false, pa.bf, false, b.bf, (short)0, o[u], false, false);
        }
    }

    // normalize and scatter into A2 columns [0, 3072)
    #pragma unroll
    for (int u = 0; u < 8; ++u)
        #pragma unroll
        for (int r = 0; r < 8; ++r) {
            int m = qrow0 + r + 8 * h2;
            int dcol = head * DHEAD + u * 16 + ln;
            a2[(size_t)m * K2 + dcol] = f32_to_bf16_bits(o[u][r] / rsum[r]);
        }
}

// ============================================================
// Kernel 6: out = A2 @ w2 + b2 ; final residual+gate epilogue
// ============================================================
__global__ void gemm2_kernel(const unsigned short* __restrict__ a2,
                             const float* __restrict__ w2,
                             const float* __restrict__ b2,
                             const float* __restrict__ x,
                             const float* __restrict__ mod,
                             float* __restrict__ out) {
    __shared__ unsigned short lA[128 * 32];
    __shared__ unsigned short lB[128 * 32];
    const int m0 = blockIdx.y * 128, n0 = blockIdx.x * 128;
    f32x8 acc[2][4];
    gemm_core(a2, K2, w2, HID, K2, L_SEQ, m0, n0, lA, lB, acc);

    const int lane = threadIdx.x & 31, w = threadIdx.x >> 5;
    const int wm = w & 3, wn = w >> 2, h2 = lane >> 4, ln = lane & 15;
    #pragma unroll
    for (int t = 0; t < 2; ++t)
        #pragma unroll
        for (int u = 0; u < 4; ++u)
            #pragma unroll
            for (int r = 0; r < 8; ++r) {
                int m = m0 + wm * 32 + t * 16 + r + 8 * h2;
                int n = n0 + wn * 64 + u * 16 + ln;
                float v = acc[t][u][r] + b2[n];
                size_t idx = (size_t)m * HID + n;
                out[idx] = x[idx] + mod[2 * HID + n] * v;
            }
}

// ============================================================
extern "C" void kernel_launch(void* const* d_in, const int* in_sizes, int n_in,
                              void* d_out, int out_size, void* d_ws, size_t ws_size,
                              hipStream_t stream) {
    const float* x        = (const float*)d_in[0];
    const float* vec      = (const float*)d_in[1];
    const float* pe       = (const float*)d_in[2];
    const float* mod_w    = (const float*)d_in[3];
    const float* mod_b    = (const float*)d_in[4];
    const float* ln_gamma = (const float*)d_in[5];
    const float* ln_beta  = (const float*)d_in[6];
    const float* w1       = (const float*)d_in[7];
    const float* b1       = (const float*)d_in[8];
    const float* q_scale  = (const float*)d_in[9];
    const float* k_scale  = (const float*)d_in[10];
    const float* w2       = (const float*)d_in[11];
    const float* b2       = (const float*)d_in[12];
    float* out = (float*)d_out;

    // workspace layout (256B aligned)
    char* ws = (char*)d_ws;
    float*          mod  = (float*)(ws);                                   // 9216 f32
    unsigned short* xmod = (unsigned short*)(ws + 40960);                  // 2048x3072 bf16
    float*          qkv  = (float*)(ws + 40960 + 12582912);                // 2048x9216 f32
    unsigned short* qb   = (unsigned short*)(ws + 40960 + 12582912 + 75497472);
    unsigned short* kb   = qb + (size_t)NHEAD * L_SEQ * DHEAD;
    unsigned short* vb   = kb + (size_t)NHEAD * L_SEQ * DHEAD;
    unsigned short* a2   = vb + (size_t)NHEAD * L_SEQ * DHEAD;             // 2048x15360 bf16

    mod_gemv_kernel<<<36, 256, 0, stream>>>(vec, mod_w, mod_b, mod);
    ln_mod_kernel<<<L_SEQ, 256, 0, stream>>>(x, mod, ln_gamma, ln_beta, xmod);
    gemm1_kernel<<<dim3(N1 / 128, L_SEQ / 128), 256, 0, stream>>>(xmod, w1, b1, qkv, a2);
    qkv_prep_kernel<<<(NHEAD * L_SEQ) / 8, 256, 0, stream>>>(qkv, pe, q_scale, k_scale, qb, kb, vb);
    attn_kernel<<<NHEAD * 16, 256, 0, stream>>>(qb, kb, vb, a2);
    gemm2_kernel<<<dim3(HID / 128, L_SEQ / 128), 256, 0, stream>>>(a2, w2, b2, x, mod, out);
}
